// GraphEncoder_65317862637631
// MI455X (gfx1250) — compile-verified, matrix operand order checked by reading, not run on hardware
//
#include <hip/hip_runtime.h>
#include <stdint.h>

// ---------------------------------------------------------------------------
// MI455X (gfx1250) implementation.  All GEMMs use v_wmma_f32_16x16x32_bf16
// (bf16 in / f32 accumulate).  Design notes:
//  * total ~100 GFLOP, working set ~150MB -> L2 (192MB) resident, compute bound
//  * GEMM K-tiles are double-buffered through LDS using the CDNA5 async
//    global->LDS DMA path (GLOBAL_LOAD_ASYNC_TO_LDS_B128 + s_wait_asynccnt),
//    overlapping the next tile's fill with the current tile's WMMAs
//  * masked neighbor-sum fused BEFORE the w_ne GEMM (linearity), bias folded
//    into the weight matrix via a count column
//  * attention projection done once per iteration; gathers read the projection
//  * all 16 gate matmuls fused into one [4096,2048]x[2048,2048] WMMA GEMM
//  * LSTM recurrence: one persistent workgroup per direction, 8 waves, h kept
//    in LDS in WMMA A-fragment layout, Whh streamed from L2 each step
// Scratch requirement: ~124 MB of d_ws.
// ---------------------------------------------------------------------------

typedef __bf16 bf16;
typedef __attribute__((ext_vector_type(16))) bf16  v16bf;
typedef __attribute__((ext_vector_type(8)))  float v8f;

union FragBF { v16bf v; uint4 q[2]; };

__device__ __forceinline__ unsigned short f2b(float f) {
  union { float f; unsigned u; } x; x.f = f;
  unsigned r = x.u + 0x7FFFu + ((x.u >> 16) & 1u);   // round-to-nearest-even
  return (unsigned short)(r >> 16);
}
__device__ __forceinline__ float sigf(float x) { return 1.0f / (1.0f + __expf(-x)); }
__device__ __forceinline__ float wred(float v) {
#pragma unroll
  for (int off = 16; off > 0; off >>= 1) v += __shfl_xor(v, off, 32);
  return v;
}
// CDNA5 async global->LDS DMA (16B per lane); tracked by ASYNCcnt.
// ldsByteOff = wave-relative LDS byte address (low 32 bits of generic addr).
__device__ __forceinline__ void ldsAsyncB128(unsigned ldsByteOff, const void* g) {
  asm volatile("global_load_async_to_lds_b128 %0, %1, off"
               :: "v"(ldsByteOff), "v"(g) : "memory");
}

// ---------------------------------------------------------------------------
// Weight pack: f32 [rows,K] -> bf16 [rows, Kpad] (dstStride), zero padded,
// optional bias column (biasCol) carrying bias[r] (used to fold b_ne).
// ---------------------------------------------------------------------------
__global__ void k_pack_w(const float* __restrict__ src, unsigned short* __restrict__ dst,
                         int rows, int K, int Kpad, int dstStride,
                         const float* __restrict__ bias, int biasCol) {
  int i = blockIdx.x * blockDim.x + threadIdx.x;
  int total = rows * Kpad;
  if (i >= total) return;
  int r = i / Kpad, k = i - r * Kpad;
  float v = 0.f;
  if (bias && k == biasCol)      v = bias[r];
  else if (k < K)                v = src[(size_t)r * K + k];
  dst[(size_t)r * dstStride + k] = f2b(v);
}

// ---------------------------------------------------------------------------
// build x = [elmo | charCNN | pos_emb | 0pad] as bf16 rows (s*8+b) x 1120.
// ---------------------------------------------------------------------------
__global__ __launch_bounds__(128) void k_build_x(
    const float* __restrict__ elmo, const int* __restrict__ pos_ids,
    const int* __restrict__ char_ids, const float* __restrict__ char_emb,
    const float* __restrict__ pos_emb, const float* __restrict__ conv_w,
    const float* __restrict__ conv_b, unsigned short* __restrict__ Xb) {
  int row = blockIdx.x;           // s*8 + b
  int s = row >> 3, b = row & 7;
  int tok = b * 512 + s;
  __shared__ float ce[16 * 30];
  __shared__ float cw[30 * 30 * 5];
  __shared__ float cf[30];
  int tid = threadIdx.x;
  for (int i = tid; i < 16 * 30; i += 128) {
    int l = i / 30, c = i - l * 30;
    ce[i] = char_emb[char_ids[tok * 16 + l] * 30 + c];
  }
  for (int i = tid; i < 4500; i += 128) cw[i] = conv_w[i];
  __syncthreads();
  if (tid < 30) {
    float m = -1e30f;
    for (int p = 0; p < 12; ++p) {        // L - K + 1 = 12 positions
      float acc = conv_b[tid];
      for (int c = 0; c < 30; ++c)
#pragma unroll
        for (int i2 = 0; i2 < 5; ++i2)
          acc += cw[tid * 150 + c * 5 + i2] * ce[(p + i2) * 30 + c];
      acc = tanhf(acc);
      m = fmaxf(m, acc);
    }
    cf[tid] = m;
  }
  __syncthreads();
  int pid = pos_ids[tok];
  for (int j = tid; j < 1120; j += 128) {
    float v;
    if (j < 1024)      v = elmo[(size_t)tok * 1024 + j];
    else if (j < 1054) v = cf[j - 1024];
    else if (j < 1104) v = pos_emb[pid * 50 + (j - 1054)];
    else               v = 0.f;
    Xb[(size_t)row * 1120 + j] = f2b(v);
  }
}

// ---------------------------------------------------------------------------
// Tiled WMMA GEMM:  C[M,N] = A[M,K] @ W[N,K]^T
// A, W bf16 row-major (K contiguous, lda/ldw multiples of 8, K mult of 32,
// M mult of 128, N mult of 64).  8 waves, each owns a 16-row strip x 64 cols.
// K-tiles double-buffered via async global->LDS DMA (ASYNCcnt).
// Fragment layout per CDNA5 ISA 7.12.2 (16-bit A 16x32; B mirrored).
// ---------------------------------------------------------------------------
#define LDT 40   // padded LDS row stride in bf16 elements (80B, 16B aligned)
__global__ __launch_bounds__(256) void k_gemm(
    const unsigned short* __restrict__ A, int lda,
    const unsigned short* __restrict__ W, int ldw,
    float* __restrict__ Cf, int ldcf,
    unsigned short* __restrict__ Cb, int ldcb, int cbOff,
    int M, int N, int K) {
  __shared__ __align__(16) unsigned short At[2][128 * LDT];
  __shared__ __align__(16) unsigned short Wt[2][64 * LDT];
  int tid = threadIdx.x, wid = tid >> 5, lane = tid & 31;
  int m0 = blockIdx.y * 128, n0 = blockIdx.x * 64;
  int row16 = lane & 15, kb = (lane >> 4) * 8;
  v8f acc[4];
#pragma unroll
  for (int nt = 0; nt < 4; ++nt)
#pragma unroll
    for (int r = 0; r < 8; ++r) acc[nt][r] = 0.f;

  int sr = tid >> 2, cg = tid & 3;
  unsigned atBase = (unsigned)(size_t)(const void*)&At[0][0];
  unsigned wtBase = (unsigned)(size_t)(const void*)&Wt[0][0];
  const unsigned atSz = 128 * LDT * 2, wtSz = 64 * LDT * 2;
  unsigned srOff = (unsigned)(sr * LDT * 2 + cg * 16);

  // issue one K-tile (3 async b128 ops per thread, completes in order)
  auto stage = [&](int buf, int kk) {
    unsigned ao = atBase + buf * atSz + srOff;
    ldsAsyncB128(ao, A + (size_t)(m0 + sr) * lda + kk + cg * 8);
    ldsAsyncB128(ao + 64u * LDT * 2, A + (size_t)(m0 + sr + 64) * lda + kk + cg * 8);
    ldsAsyncB128(wtBase + buf * wtSz + srOff, W + (size_t)(n0 + sr) * ldw + kk + cg * 8);
    if (kk + 64 < K)   // warm L2 two tiles ahead
      __builtin_prefetch(A + (size_t)(m0 + sr) * lda + kk + 64 + cg * 8, 0, 0);
  };

  stage(0, 0);
  int nk = K >> 5;
  for (int i = 0; i < nk; ++i) {
    int cur = i & 1;
    bool more = (i + 1) < nk;
    if (more) {
      stage(cur ^ 1, (i + 1) << 5);
      // 3 in-flight ops belong to the next stage; in-order completion =>
      // everything for the current stage has landed in LDS.
      asm volatile("s_wait_asynccnt 0x3" ::: "memory");
    } else {
      asm volatile("s_wait_asynccnt 0x0" ::: "memory");
    }
    __syncthreads();
    const unsigned short* Ab = At[cur];
    const unsigned short* Wb = Wt[cur];
    FragBF a;
    const unsigned short* ap = Ab + (wid * 16 + row16) * LDT + kb;
    a.q[0] = *(const uint4*)ap;
    a.q[1] = *(const uint4*)(ap + 16);
#pragma unroll
    for (int nt = 0; nt < 4; ++nt) {
      FragBF bm;
      const unsigned short* bp = Wb + (nt * 16 + row16) * LDT + kb;
      bm.q[0] = *(const uint4*)bp;
      bm.q[1] = *(const uint4*)(bp + 16);
      acc[nt] = __builtin_amdgcn_wmma_f32_16x16x32_bf16(
          false, a.v, false, bm.v, (short)0, acc[nt], false, false);
    }
    __syncthreads();   // all reads of buf[cur] done before it is re-staged
  }
  int mr = m0 + wid * 16 + (lane >> 4) * 8;
#pragma unroll
  for (int nt = 0; nt < 4; ++nt) {
    int n = n0 + nt * 16 + row16;
#pragma unroll
    for (int r = 0; r < 8; ++r) {
      float v = acc[nt][r];
      if (Cf) Cf[(size_t)(mr + r) * ldcf + n] = v;
      if (Cb) Cb[(size_t)(mr + r) * ldcb + cbOff + n] = f2b(v);
    }
  }
}

// ---------------------------------------------------------------------------
// BiLSTM recurrence.  grid=2 (dir), 256 threads = 8 waves; wave w covers
// N in [w*128, w*128+128).  h kept in LDS bf16 (rows 8..15 zero), Whh read
// from L2 each step.  Writes lstm_out rows b*512+t (f32 + bf16 copies).
// ---------------------------------------------------------------------------
__global__ __launch_bounds__(256) void k_lstm(
    const unsigned short* __restrict__ WhhF, const unsigned short* __restrict__ WhhB,
    const float* __restrict__ Zx,      // [512*8, 2048], cols dir*1024..
    const float* __restrict__ biasF, const float* __restrict__ biasB,
    float* __restrict__ LO, unsigned short* __restrict__ LOb, int S) {
  int dir = blockIdx.x;
  const unsigned short* Whh = dir ? WhhB : WhhF;
  const float* bias = dir ? biasB : biasF;
  int zcol = dir * 1024, ocol = dir * 256;
  __shared__ __align__(16) unsigned short hbf[16 * 264];  // 264 = 256 + 8 pad
  __shared__ float zbuf[8 * 1024];
  int tid = threadIdx.x, wid = tid >> 5, lane = tid & 31;
  int row16 = lane & 15, kb = (lane >> 4) * 8;
  for (int i = tid; i < 16 * 264; i += 256) hbf[i] = 0;
  float c[8];
#pragma unroll
  for (int i = 0; i < 8; ++i) c[i] = 0.f;
  __syncthreads();

  for (int step = 0; step < S; ++step) {
    int t = dir ? (S - 1 - step) : step;
    v8f acc[8];
#pragma unroll
    for (int tt = 0; tt < 8; ++tt)
#pragma unroll
      for (int r = 0; r < 8; ++r) acc[tt][r] = 0.f;
    for (int k0 = 0; k0 < 256; k0 += 32) {
      FragBF a;
      const unsigned short* ap = hbf + row16 * 264 + k0 + kb;
      a.q[0] = *(const uint4*)ap;
      a.q[1] = *(const uint4*)(ap + 16);
#pragma unroll
      for (int tt = 0; tt < 8; ++tt) {
        int n = wid * 128 + tt * 16 + row16;
        FragBF wm;
        const unsigned short* wp = Whh + (size_t)n * 256 + k0 + kb;
        wm.q[0] = *(const uint4*)wp;
        wm.q[1] = *(const uint4*)(wp + 16);
        acc[tt] = __builtin_amdgcn_wmma_f32_16x16x32_bf16(
            false, a.v, false, wm.v, (short)0, acc[tt], false, false);
      }
    }
    if (lane < 16) {  // lanes 0..15 hold C rows 0..7 (the 8 valid batch rows)
#pragma unroll
      for (int tt = 0; tt < 8; ++tt) {
        int nl = wid * 128 + tt * 16 + row16;
#pragma unroll
        for (int r = 0; r < 8; ++r)
          zbuf[r * 1024 + nl] = acc[tt][r] + Zx[(size_t)(t * 8 + r) * 2048 + zcol + nl];
      }
    }
    __syncthreads();
    float hnew[8];
#pragma unroll
    for (int i = 0; i < 8; ++i) {
      int e = i * 256 + tid;
      int b = e >> 8, j = e & 255;
      float zi = zbuf[b * 1024 + j]       + bias[j];
      float zf = zbuf[b * 1024 + 256 + j] + bias[256 + j];
      float zg = zbuf[b * 1024 + 512 + j] + bias[512 + j];
      float zo = zbuf[b * 1024 + 768 + j] + bias[768 + j];
      float ci = sigf(zf) * c[i] + sigf(zi) * tanhf(zg);
      c[i] = ci;
      float h = sigf(zo) * tanhf(ci);
      hnew[i] = h;
      size_t orow = (size_t)b * S + t;
      LO [orow * 512 + ocol + j] = h;
      LOb[orow * 512 + ocol + j] = f2b(h);
    }
#pragma unroll
    for (int i = 0; i < 8; ++i) {
      int e = i * 256 + tid;
      int b = e >> 8, j = e & 255;
      hbf[b * 264 + j] = f2b(hnew[i]);
    }
    __syncthreads();
  }
}

// ---------------------------------------------------------------------------
// Masked neighbor-sum (fused pre-GEMM):  G[r, 0:512]  = sum_n m*LO[gather]
//                                        G[r,512:612] = sum_n m*edge_emb
//                                        G[r,612]     = sum_n m   (bias count)
// ---------------------------------------------------------------------------
__global__ __launch_bounds__(128) void k_gather_ne(
    const float* __restrict__ LO, const int* __restrict__ idx,
    const int* __restrict__ mask, const int* __restrict__ eidx,
    const float* __restrict__ edge_emb, unsigned short* __restrict__ G, int S) {
  int r = blockIdx.x;            // b*S + s
  int b = r / S;
  __shared__ int sI[8], sM[8], sE[8];
  if (threadIdx.x < 8) {
    sI[threadIdx.x] = idx[r * 8 + threadIdx.x];
    sM[threadIdx.x] = mask[r * 8 + threadIdx.x];
    sE[threadIdx.x] = eidx[r * 8 + threadIdx.x];
  }
  __syncthreads();
  for (int col = threadIdx.x; col < 640; col += 128) {
    float acc = 0.f;
    if (col < 512) {
#pragma unroll
      for (int n = 0; n < 8; ++n)
        if (sM[n]) acc += LO[((size_t)b * S + sI[n]) * 512 + col];
    } else if (col < 612) {
#pragma unroll
      for (int n = 0; n < 8; ++n)
        if (sM[n]) acc += edge_emb[sE[n] * 100 + (col - 512)];
    } else if (col == 612) {
#pragma unroll
      for (int n = 0; n < 8; ++n) acc += (float)sM[n];
    }
    G[(size_t)r * 640 + col] = f2b(acc);
  }
}

// ---------------------------------------------------------------------------
// Graph attention for one direction.  One wave per (token, head).
// P = h @ attn_w^T (projected once); gather commutes with projection.
// ---------------------------------------------------------------------------
__global__ __launch_bounds__(256) void k_attn(
    const float* __restrict__ P, const int* __restrict__ idx,
    const int* __restrict__ mask, const float* __restrict__ avec,
    unsigned short* __restrict__ X4, int colOff, int S) {
  int r = blockIdx.x;            // b*S + s
  int b = r / S;
  int tid = threadIdx.x, h = tid >> 5, lane = tid & 31;
  int d0 = lane * 2, d1 = d0 + 1;
  float a1_0 = avec[h * 128 + d0],      a1_1 = avec[h * 128 + d1];
  float a2_0 = avec[h * 128 + 64 + d0], a2_1 = avec[h * 128 + 64 + d1];
  float pn0 = P[(size_t)r * 512 + h * 64 + d0];
  float pn1 = P[(size_t)r * 512 + h * 64 + d1];
  float nodedot = wred(a1_0 * pn0 + a1_1 * pn1);
  float v0[8], v1[8], sc[8];
#pragma unroll
  for (int n = 0; n < 8; ++n) {
    int nr = idx[r * 8 + n];
    size_t rn = (size_t)b * S + nr;
    v0[n] = P[rn * 512 + h * 64 + d0];
    v1[n] = P[rn * 512 + h * 64 + d1];
    float s = nodedot + wred(a2_0 * v0[n] + a2_1 * v1[n]);
    s = (s > 0.f) ? s : 0.01f * s;                // leaky relu
    if (mask[r * 8 + n] == 0) s = -1e10f;
    sc[n] = s;
  }
  float mx = sc[0];
#pragma unroll
  for (int n = 1; n < 8; ++n) mx = fmaxf(mx, sc[n]);
  float se = 0.f;
#pragma unroll
  for (int n = 0; n < 8; ++n) { sc[n] = __expf(sc[n] - mx); se += sc[n]; }
  float inv = 1.f / se;
  float o0 = 0.f, o1 = 0.f;
#pragma unroll
  for (int n = 0; n < 8; ++n) { o0 += sc[n] * v0[n]; o1 += sc[n] * v1[n]; }
  X4[(size_t)r * 2048 + colOff + h * 64 + d0] = f2b(o0 * inv);
  X4[(size_t)r * 2048 + colOff + h * 64 + d1] = f2b(o1 * inv);
}

// ---------------------------------------------------------------------------
// Gate elementwise: Zg cols [i|f|o|g] blocks of 512.
// ---------------------------------------------------------------------------
__global__ void k_gates(const float* __restrict__ Zg, const float* __restrict__ bi,
                        const float* __restrict__ bff, const float* __restrict__ bo,
                        const float* __restrict__ bg, float* __restrict__ C,
                        float* __restrict__ H, unsigned short* __restrict__ Hb) {
  int e = blockIdx.x * blockDim.x + threadIdx.x;
  if (e >= 4096 * 512) return;
  int row = e >> 9, j = e & 511;
  const float* z = Zg + (size_t)row * 2048;
  float i = sigf(z[j] + bi[j]);
  float f = sigf(z[512 + j] + bff[j]);
  float o = sigf(z[1024 + j] + bo[j]);
  float g = tanhf(z[1536 + j] + bg[j]);
  float c = f * C[e] + i * g;
  C[e] = c;
  float h = o * tanhf(c);
  H[e] = h;
  Hb[e] = f2b(h);
}

__global__ void k_copy(const float* __restrict__ s, float* __restrict__ d, int n) {
  int i = blockIdx.x * blockDim.x + threadIdx.x;
  if (i < n) d[i] = s[i];
}

// ---------------------------------------------------------------------------
extern "C" void kernel_launch(void* const* d_in, const int* in_sizes, int n_in,
                              void* d_out, int out_size, void* d_ws, size_t ws_size,
                              hipStream_t stream) {
  (void)in_sizes; (void)n_in; (void)out_size; (void)ws_size;
  const float* elmo     = (const float*)d_in[0];
  const int*   pos_ids  = (const int*)d_in[2];
  const int*   char_ids = (const int*)d_in[3];
  const int*   in_idx   = (const int*)d_in[4];
  const int*   in_mask  = (const int*)d_in[5];
  const int*   out_idx  = (const int*)d_in[6];
  const int*   out_mask = (const int*)d_in[7];
  const int*   in_eidx  = (const int*)d_in[8];
  const int*   out_eidx = (const int*)d_in[9];
  const float* char_emb = (const float*)d_in[10];
  const float* pos_emb  = (const float*)d_in[11];
  const float* edge_emb = (const float*)d_in[12];
  const float* conv_w   = (const float*)d_in[13];
  const float* conv_b   = (const float*)d_in[14];
  const float* wih_f    = (const float*)d_in[15];
  const float* whh_f    = (const float*)d_in[16];
  const float* b_f      = (const float*)d_in[17];
  const float* wih_b    = (const float*)d_in[18];
  const float* whh_b    = (const float*)d_in[19];
  const float* b_b      = (const float*)d_in[20];
  const float* w_ne     = (const float*)d_in[21];
  const float* b_ne     = (const float*)d_in[22];
  const float* w_h      = (const float*)d_in[23];
  const float* w_cell   = (const float*)d_in[24];
  const float* aw_in    = (const float*)d_in[25];
  const float* aa_in    = (const float*)d_in[26];
  const float* aw_out   = (const float*)d_in[27];
  const float* aa_out   = (const float*)d_in[28];
  // gates: base 29 + g*5 -> w_in, u_in, w_out, u_out, b

  // ---- scratch carve-out (256B aligned) ----
  size_t cur = 0;
  auto alloc = [&](size_t bytes) -> char* {
    char* p = (char*)d_ws + cur;
    cur = (cur + bytes + 255) & ~(size_t)255;
    return p;
  };
  unsigned short* WihCat = (unsigned short*)alloc((size_t)2048 * 1120 * 2);
  unsigned short* WhhF   = (unsigned short*)alloc((size_t)1024 * 256 * 2);
  unsigned short* WhhB   = (unsigned short*)alloc((size_t)1024 * 256 * 2);
  unsigned short* WnePk  = (unsigned short*)alloc((size_t)512 * 640 * 2);
  unsigned short* Wh     = (unsigned short*)alloc((size_t)512 * 512 * 2);
  unsigned short* Wcell  = (unsigned short*)alloc((size_t)512 * 512 * 2);
  unsigned short* AWin   = (unsigned short*)alloc((size_t)512 * 512 * 2);
  unsigned short* AWout  = (unsigned short*)alloc((size_t)512 * 512 * 2);
  unsigned short* Wall   = (unsigned short*)alloc((size_t)2048 * 2048 * 2);
  unsigned short* Xb     = (unsigned short*)alloc((size_t)4096 * 1120 * 2);
  float*          Z      = (float*)alloc((size_t)4096 * 2048 * 4);   // Zx then Zg
  float*          LO     = (float*)alloc((size_t)4096 * 512 * 4);
  unsigned short* LOb    = (unsigned short*)alloc((size_t)4096 * 512 * 2);
  unsigned short* G      = (unsigned short*)alloc((size_t)4096 * 640 * 2);
  unsigned short* X4     = (unsigned short*)alloc((size_t)4096 * 2048 * 2);
  float*          H      = (float*)alloc((size_t)4096 * 512 * 4);
  float*          Cst    = (float*)alloc((size_t)4096 * 512 * 4);
  unsigned short* Hb     = (unsigned short*)alloc((size_t)4096 * 512 * 2);
  float*          P      = (float*)alloc((size_t)4096 * 512 * 4);

  auto pack = [&](const float* src, unsigned short* dst, int rows, int K, int Kpad,
                  int stride, const float* bias, int biasCol) {
    int total = rows * Kpad;
    k_pack_w<<<(total + 255) / 256, 256, 0, stream>>>(src, dst, rows, K, Kpad, stride,
                                                      bias, biasCol);
  };
  auto gemm = [&](const unsigned short* A, int lda, const unsigned short* W, int ldw,
                  float* Cf, int ldcf, unsigned short* Cb, int ldcb, int cbOff,
                  int M, int N, int K) {
    dim3 g(N / 64, M / 128);
    k_gemm<<<g, 256, 0, stream>>>(A, lda, W, ldw, Cf, ldcf, Cb, ldcb, cbOff, M, N, K);
  };

  // ---- stage 0: weight packing ----
  pack(wih_f, WihCat, 1024, 1104, 1120, 1120, nullptr, -1);
  pack(wih_b, WihCat + (size_t)1024 * 1120, 1024, 1104, 1120, 1120, nullptr, -1);
  pack(whh_f, WhhF, 1024, 256, 256, 256, nullptr, -1);
  pack(whh_b, WhhB, 1024, 256, 256, 256, nullptr, -1);
  pack(w_ne, WnePk, 512, 612, 640, 640, b_ne, 612);   // fold bias at col 612
  pack(w_h, Wh, 512, 512, 512, 512, nullptr, -1);
  pack(w_cell, Wcell, 512, 512, 512, 512, nullptr, -1);
  pack(aw_in, AWin, 512, 512, 512, 512, nullptr, -1);
  pack(aw_out, AWout, 512, 512, 512, 512, nullptr, -1);
  for (int g = 0; g < 4; ++g)
    for (int blk = 0; blk < 4; ++blk)
      pack((const float*)d_in[29 + g * 5 + blk],
           Wall + (size_t)(g * 512) * 2048 + blk * 512, 512, 512, 512, 2048,
           nullptr, -1);

  // ---- stage 1: input features ----
  k_build_x<<<4096, 128, 0, stream>>>(elmo, pos_ids, char_ids, char_emb, pos_emb,
                                      conv_w, conv_b, Xb);

  // ---- stage 2: LSTM input projection (both dirs) ----
  gemm(Xb, 1120, WihCat, 1120, Z, 2048, nullptr, 0, 0, 4096, 2048, 1120);

  // ---- stage 3: recurrence ----
  k_lstm<<<2, 256, 0, stream>>>(WhhF, WhhB, Z, b_f, b_b, LO, LOb, 512);

  // ---- stage 4: node+edge reps (fused masked sum -> GEMM -> X4 bf16) ----
  k_gather_ne<<<4096, 128, 0, stream>>>(LO, in_idx, in_mask, in_eidx, edge_emb, G, 512);
  gemm(G, 640, WnePk, 640, nullptr, 0, X4, 2048, 0, 4096, 512, 640);
  k_gather_ne<<<4096, 128, 0, stream>>>(LO, out_idx, out_mask, out_eidx, edge_emb, G, 512);
  gemm(G, 640, WnePk, 640, nullptr, 0, X4, 2048, 1024, 4096, 512, 640);

  // ---- stage 5: h0 / c0 ----
  gemm(LOb, 512, Wh, 512, H, 512, Hb, 512, 0, 4096, 512, 512);
  gemm(LOb, 512, Wcell, 512, Cst, 512, nullptr, 0, 0, 4096, 512, 512);

  // ---- stage 6: 3 message-passing iterations ----
  for (int it = 0; it < 3; ++it) {
    gemm(Hb, 512, AWin, 512, P, 512, nullptr, 0, 0, 4096, 512, 512);
    k_attn<<<4096, 256, 0, stream>>>(P, in_idx, in_mask, aa_in, X4, 512, 512);
    gemm(Hb, 512, AWout, 512, P, 512, nullptr, 0, 0, 4096, 512, 512);
    k_attn<<<4096, 256, 0, stream>>>(P, out_idx, out_mask, aa_out, X4, 1536, 512);
    gemm(X4, 2048, Wall, 2048, Z, 2048, nullptr, 0, 0, 4096, 2048, 2048);
    k_gates<<<(4096 * 512) / 256, 256, 0, stream>>>(
        Z, (const float*)d_in[33], (const float*)d_in[38],
        (const float*)d_in[43], (const float*)d_in[48], Cst, H, Hb);
  }

  // ---- output ----
  k_copy<<<(4096 * 512) / 256, 256, 0, stream>>>(H, (float*)d_out, 4096 * 512);
}
